// GRU_64793876628225
// MI455X (gfx1250) — matrix operand order repaired and checked by reading, not compiled, not run on hardware
//
#include <hip/hip_runtime.h>
#include <hip/hip_bf16.h>
#include <stdint.h>

#define B_  64
#define T_  2048
#define I_  256
#define H_  256
#define BT_ (B_ * T_)
#define BTH_ ((size_t)BT_ * H_)
#define HH_ (H_ * I_)        // 65536 elements per weight matrix

#define WPAD  264            // padded u16 row stride for bf16 LDS tiles (528B = 33*16B)
#define HFPAD 260            // padded f32 row stride for h state

typedef __attribute__((ext_vector_type(16))) __bf16       v16bf;
typedef __attribute__((ext_vector_type(8)))  float        v8f;
typedef __attribute__((ext_vector_type(4)))  unsigned int u32x4;
typedef __attribute__((ext_vector_type(8)))  int          i32x8;
typedef __attribute__((ext_vector_type(4)))  int          i32x4;

union Frag16 { v16bf v; u32x4 q[2]; };

__device__ __forceinline__ unsigned short f2bf(float f) {
    union { float f; uint32_t u; } c; c.f = f;
    uint32_t u = c.u;
    uint32_t r = (u + 0x7FFFu + ((u >> 16) & 1u)) >> 16;   // round-to-nearest-even
    return (unsigned short)r;
}

__device__ __forceinline__ v8f v8f_zero() {
    v8f z;
#pragma unroll
    for (int i = 0; i < 8; ++i) z[i] = 0.0f;
    return z;
}

__device__ __forceinline__ float fsig(float x) { return 1.0f / (1.0f + __expf(-x)); }
__device__ __forceinline__ float ftanh(float x) {
    float ax = fabsf(x);
    float e  = __expf(-2.0f * ax);
    float t  = (1.0f - e) / (1.0f + e);
    return copysignf(t, x);
}

// ---- TDM: DMA a 256x256 bf16 tile (row-major, 512B rows) into LDS with a
//      16B pad appended after every 512B row (-> 528B LDS row stride = WPAD).
//      D# built per CDNA5 ISA 8.3/8.4; 2-D tile so groups 2/3 (and the extra
//      descriptor words of the 6-arg builtin) are zero.
__device__ __forceinline__ void tdm_load_w_padded(uint32_t lds_byte_addr, const void* gsrc) {
    uint64_t ga = (uint64_t)(uintptr_t)gsrc;
    u32x4 g0;
    g0[0] = 1u;                                             // count=1 valid descriptor
    g0[1] = lds_byte_addr;                                  // lds_addr (bytes)
    g0[2] = (uint32_t)ga;                                   // global_addr[31:0]
    g0[3] = (uint32_t)((ga >> 32) & 0x1FFFFFFu) | (2u << 30); // addr[56:32] | type=2
    i32x8 g1;
    g1[0] = (int)((1u << 16)      // data_size = 1 -> 2 bytes
                | (1u << 20)      // pad_enable
                | (6u << 22)      // pad_interval: 128 DWORDs = 512B
                | (3u << 25));    // pad_amount: 4 DWORDs = 16B
    g1[1] = (int)(256u << 16);    // tensor_dim0 = 256 (lo16 @ bits 63:48)
    g1[2] = (int)(256u << 16);    // tensor_dim0 hi=0, tensor_dim1 = 256 (lo16 @ 95:80)
    g1[3] = (int)(256u << 16);    // tensor_dim1 hi=0, tile_dim0 = 256 (bits 127:112)
    g1[4] = (int)256u;            // tile_dim1 = 256 (bits 143:128), tile_dim2 = 0
    g1[5] = (int)256u;            // tensor_dim0_stride = 256 elements (bits 191:160)
    g1[6] = 0;                    // stride0 hi, tensor_dim1_stride lo = 0
    g1[7] = 0;
    i32x4 gz4;
    gz4[0] = 0; gz4[1] = 0; gz4[2] = 0; gz4[3] = 0;
    i32x8 gz8;
#pragma unroll
    for (int i = 0; i < 8; ++i) gz8[i] = 0;
    __builtin_amdgcn_tensor_load_to_lds(g0, g1, gz4, gz4, gz8, 0);
}

// ---------------- Phase 0: fp32 -> bf16 conversions ----------------
__global__ void cvt_x_bf16(const float* __restrict__ x, unsigned short* __restrict__ xb, int n) {
    int i = blockIdx.x * blockDim.x + threadIdx.x;
    int stride = gridDim.x * blockDim.x;
    for (; i < n; i += stride) xb[i] = f2bf(x[i]);
}

__global__ void cvt_w_bf16(const float* __restrict__ a, const float* __restrict__ b,
                           const float* __restrict__ c, const float* __restrict__ d,
                           const float* __restrict__ e, const float* __restrict__ f,
                           unsigned short* __restrict__ dst) {
    int i = blockIdx.x * blockDim.x + threadIdx.x;
    if (i >= HH_) return;
    dst[0 * HH_ + i] = f2bf(a[i]);   // Wir
    dst[1 * HH_ + i] = f2bf(b[i]);   // Wiz
    dst[2 * HH_ + i] = f2bf(c[i]);   // WiN
    dst[3 * HH_ + i] = f2bf(d[i]);   // Whr
    dst[4 * HH_ + i] = f2bf(e[i]);   // Whz
    dst[5 * HH_ + i] = f2bf(f[i]);   // WhN
}

// ---------------- Phase 1: input projections (massively parallel WMMA GEMM) ----------------
// One gate, one 16-row M tile per wave, N processed in two halves of 8 tiles so
// acc[8]+b[8] (~140 VGPRs) stays inside the direct 256-VGPR window: the
// scheduler can keep every B fragment live and drain loads with partial
// s_wait_loadcnt instead of per-tile full stalls.
__device__ __forceinline__ void xproj_gate(const unsigned short* __restrict__ xb,
                                           const unsigned short* __restrict__ W,
                                           const float* __restrict__ bias,
                                           float* __restrict__ op,
                                           int mbase, int half, int l16) {
    const size_t arow = (size_t)(mbase + l16);
#pragma unroll
    for (int h2 = 0; h2 < 2; ++h2) {          // N half: tiles 8*h2 .. 8*h2+7
        v8f acc[8];
#pragma unroll
        for (int q = 0; q < 8; ++q) acc[q] = v8f_zero();

        for (int kt = 0; kt < 8; ++kt) {
            Frag16 a;   // A: 16x32 bf16, row = lane%16, K = 32*kt + 16*half + j
            const unsigned short* ap = xb + arow * I_ + kt * 32 + half * 16;
            a.q[0] = *reinterpret_cast<const u32x4*>(ap);
            a.q[1] = *reinterpret_cast<const u32x4*>(ap + 8);

            Frag16 b[8];   // B: 32x16 bf16, col = lane%16, K = 32*kt + 16*half + j
#pragma unroll
            for (int q = 0; q < 8; ++q) {
                const int nt = 8 * h2 + q;
                const unsigned short* bp = W + (size_t)(nt * 16 + l16) * I_ + kt * 32 + half * 16;
                b[q].q[0] = *reinterpret_cast<const u32x4*>(bp);
                b[q].q[1] = *reinterpret_cast<const u32x4*>(bp + 8);
            }
#pragma unroll
            for (int q = 0; q < 8; ++q) {
                acc[q] = __builtin_amdgcn_wmma_f32_16x16x32_bf16(
                    false, a.v, false, b[q].v, (short)0, acc[q], false, false);
            }
        }
#pragma unroll
        for (int q = 0; q < 8; ++q) {
#pragma unroll
            for (int i = 0; i < 8; ++i) {
                int m = i + half * 8;
                int n = (8 * h2 + q) * 16 + l16;
                op[(size_t)(mbase + m) * H_ + n] = acc[q][i] + bias[n];
            }
        }
    }
}

__global__ __launch_bounds__(256)
void gru_xproj(const unsigned short* __restrict__ xb,   // [BT, I] bf16
               const unsigned short* __restrict__ wiB,  // [3][H, I] bf16
               const float* __restrict__ bir, const float* __restrict__ biz,
               const float* __restrict__ biN,
               float* __restrict__ xr, float* __restrict__ xz, float* __restrict__ xn) {
    const int lane  = threadIdx.x & 31;
    const int wv    = threadIdx.x >> 5;
    const int half  = lane >> 4;
    const int l16   = lane & 15;
    const int mbase = blockIdx.x * 128 + wv * 16;

    xproj_gate(xb, wiB + 0 * (size_t)HH_, bir, xr, mbase, half, l16);
    xproj_gate(xb, wiB + 1 * (size_t)HH_, biz, xz, mbase, half, l16);
    xproj_gate(xb, wiB + 2 * (size_t)HH_, biN, xn, mbase, half, l16);
}

// ---------------- Phase 2: sequential scan (4 WGs x 16 batch rows, T=2048 steps) ----------------
// Whr/Whz DMA'd into padded LDS by the Tensor Data Mover; WhN fragments live in
// VGPRs; h kept in LDS as bf16 (A fragments) + f32 (state).
__global__ __launch_bounds__(256)
void gru_scan(const unsigned short* __restrict__ whB,   // [3][H, H] bf16: Whr, Whz, WhN
              const float* __restrict__ bhr, const float* __restrict__ bhz,
              const float* __restrict__ bhN,
              const float* __restrict__ xr, const float* __restrict__ xz,
              const float* __restrict__ xn,
              float* __restrict__ out, float* __restrict__ hlast) {
    extern __shared__ char smem[];
    unsigned short* whrL = (unsigned short*)smem;          // 256 * WPAD u16
    unsigned short* whzL = whrL + 256 * WPAD;              // 256 * WPAD u16
    unsigned short* hbf  = whzL + 256 * WPAD;              // 16  * WPAD u16
    float*          hf   = (float*)(hbf + 16 * WPAD);      // 16  * HFPAD f32

    const int tid  = threadIdx.x;
    const int lane = tid & 31;
    const int wv   = tid >> 5;
    const int half = lane >> 4;
    const int l16  = lane & 15;
    const int b0   = blockIdx.x * 16;

    // TDM: one DMA per weight matrix, issued by waves 0 and 1 (EXEC-independent,
    // tracked with TENSORcnt). Pad fields reproduce the WPAD LDS layout.
    if (wv == 0) tdm_load_w_padded((uint32_t)(uintptr_t)whrL, whB);
    if (wv == 1) tdm_load_w_padded((uint32_t)(uintptr_t)whzL, whB + HH_);

    // Zero h state while the DMA streams in.
    for (int idx = tid; idx < 16 * WPAD; idx += 256)  hbf[idx] = 0;
    for (int idx = tid; idx < 16 * HFPAD; idx += 256) hf[idx]  = 0.0f;

    // Hoist WhN fragments into registers (2 N-tiles per wave x 8 K-steps = 128 VGPRs).
    Frag16 bN[2][8];
#pragma unroll
    for (int u = 0; u < 2; ++u) {
        int n0 = (2 * wv + u) * 16;
#pragma unroll
        for (int kt = 0; kt < 8; ++kt) {
            const unsigned short* bp = whB + 2 * HH_ + (size_t)(n0 + l16) * H_ + kt * 32 + half * 16;
            bN[u][kt].q[0] = *reinterpret_cast<const u32x4*>(bp);
            bN[u][kt].q[1] = *reinterpret_cast<const u32x4*>(bp + 8);
        }
    }
    // Per-lane recurrent biases.
    float bR[2], bZ[2], bNN[2];
#pragma unroll
    for (int u = 0; u < 2; ++u) {
        int n = (2 * wv + u) * 16 + l16;
        bR[u] = bhr[n]; bZ[u] = bhz[n]; bNN[u] = bhN[n];
    }

    __builtin_amdgcn_s_wait_tensorcnt(0);   // waves 0/1 drain their DMA; others pass
    __syncthreads();

    for (int t = 0; t < T_; ++t) {
        // h A-fragments for this step: loaded once, reused by 2 N-tiles x 3 gates.
        Frag16 af[8];
#pragma unroll
        for (int kt = 0; kt < 8; ++kt) {
            const unsigned short* ap = hbf + l16 * WPAD + kt * 32 + half * 16;
            af[kt].q[0] = *reinterpret_cast<const u32x4*>(ap);
            af[kt].q[1] = *reinterpret_cast<const u32x4*>(ap + 8);
        }

        float hnew[2][8];
#pragma unroll
        for (int u = 0; u < 2; ++u) {
            const int nt = 2 * wv + u, n0 = nt * 16;
            v8f aR = v8f_zero(), aZ = v8f_zero(), aN = v8f_zero();
#pragma unroll
            for (int kt = 0; kt < 8; ++kt) {
                Frag16 br, bz;
                const unsigned short* rp = whrL + (n0 + l16) * WPAD + kt * 32 + half * 16;
                br.q[0] = *reinterpret_cast<const u32x4*>(rp);
                br.q[1] = *reinterpret_cast<const u32x4*>(rp + 8);
                const unsigned short* zp = whzL + (n0 + l16) * WPAD + kt * 32 + half * 16;
                bz.q[0] = *reinterpret_cast<const u32x4*>(zp);
                bz.q[1] = *reinterpret_cast<const u32x4*>(zp + 8);
                aR = __builtin_amdgcn_wmma_f32_16x16x32_bf16(false, af[kt].v, false, br.v,        (short)0, aR, false, false);
                aZ = __builtin_amdgcn_wmma_f32_16x16x32_bf16(false, af[kt].v, false, bz.v,        (short)0, aZ, false, false);
                aN = __builtin_amdgcn_wmma_f32_16x16x32_bf16(false, af[kt].v, false, bN[u][kt].v, (short)0, aN, false, false);
            }
#pragma unroll
            for (int i = 0; i < 8; ++i) {
                int m = i + half * 8;
                int n = n0 + l16;
                size_t grow = (size_t)(b0 + m) * T_ + t;
                float xrv = xr[grow * H_ + n];
                float xzv = xz[grow * H_ + n];
                float xnv = xn[grow * H_ + n];
                float r  = fsig(xrv + aR[i] + bR[u]);
                float z  = fsig(xzv + aZ[i] + bZ[u]);
                float nn = ftanh(xnv + r * (aN[i] + bNN[u]));
                float ho = hf[m * HFPAD + n];
                float hv = (1.0f - z) * nn + z * ho;
                hnew[u][i] = hv;
                out[grow * H_ + n] = hv;
                // Hide next step's gate-input latency (global_prefetch_b8).
                if (i == 0 && t + 1 < T_) {
                    __builtin_prefetch(&xr[grow * H_ + H_ + n], 0, 3);
                    __builtin_prefetch(&xz[grow * H_ + H_ + n], 0, 3);
                    __builtin_prefetch(&xn[grow * H_ + H_ + n], 0, 3);
                }
            }
        }
        __syncthreads();   // everyone done reading old h
#pragma unroll
        for (int u = 0; u < 2; ++u) {
            int n = (2 * wv + u) * 16 + l16;
#pragma unroll
            for (int i = 0; i < 8; ++i) {
                int m = i + half * 8;
                hf[m * HFPAD + n]  = hnew[u][i];
                hbf[m * WPAD + n]  = f2bf(hnew[u][i]);
            }
        }
        __syncthreads();   // new h visible before next step
    }

    for (int idx = tid; idx < 16 * H_; idx += 256) {
        int m = idx >> 8, n = idx & 255;
        hlast[(size_t)(b0 + m) * H_ + n] = hf[m * HFPAD + n];
    }
}

// ---------------- Launch ----------------
extern "C" void kernel_launch(void* const* d_in, const int* in_sizes, int n_in,
                              void* d_out, int out_size, void* d_ws, size_t ws_size,
                              hipStream_t stream) {
    const float* X   = (const float*)d_in[0];
    const float* Wir = (const float*)d_in[1];
    const float* bir = (const float*)d_in[2];
    const float* Whr = (const float*)d_in[3];
    const float* bhr = (const float*)d_in[4];
    const float* Wiz = (const float*)d_in[5];
    const float* biz = (const float*)d_in[6];
    const float* Whz = (const float*)d_in[7];
    const float* bhz = (const float*)d_in[8];
    const float* WiN = (const float*)d_in[9];
    const float* biN = (const float*)d_in[10];
    const float* WhN = (const float*)d_in[11];
    const float* bhN = (const float*)d_in[12];

    (void)in_sizes; (void)n_in; (void)out_size; (void)ws_size;

    // Workspace layout: [6 bf16 weight matrices][xr f32][xz f32][xn f32][X bf16]
    char* ws = (char*)d_ws;
    unsigned short* wiB = (unsigned short*)ws;              // [Wir,Wiz,WiN]
    unsigned short* whB = wiB + 3 * HH_;                    // [Whr,Whz,WhN]
    float* xr = (float*)(ws + (size_t)6 * HH_ * sizeof(unsigned short));
    float* xz = xr + BTH_;
    float* xn = xz + BTH_;
    unsigned short* xb = (unsigned short*)(xn + BTH_);

    float* out   = (float*)d_out;
    float* hlast = out + BTH_;

    cvt_x_bf16<<<2048, 256, 0, stream>>>(X, xb, BT_ * I_);
    cvt_w_bf16<<<(HH_ + 255) / 256, 256, 0, stream>>>(Wir, Wiz, WiN, Whr, Whz, WhN, wiB);
    gru_xproj<<<BT_ / 128, 256, 0, stream>>>(xb, wiB, bir, biz, biN, xr, xz, xn);

    size_t smem = (size_t)(2 * 256 * WPAD + 16 * WPAD) * sizeof(unsigned short)
                + (size_t)(16 * HFPAD) * sizeof(float);    // ~295 KB of the 320 KB WGP LDS
    gru_scan<<<4, 256, smem, stream>>>(whB, bhr, bhz, bhN, xr, xz, xn, out, hlast);
}
